// ColOutlierLinear_89369679495623
// MI455X (gfx1250) — matrix operand order repaired and checked by reading, not compiled, not run on hardware
//
#include <hip/hip_runtime.h>
#include <stdint.h>

typedef __attribute__((ext_vector_type(16))) _Float16 v16h;
typedef __attribute__((ext_vector_type(8)))  _Float16 v8h;
typedef __attribute__((ext_vector_type(4)))  _Float16 v4h;
typedef __attribute__((ext_vector_type(8)))  float    v8f;
typedef __attribute__((ext_vector_type(4)))  float    v4f;
typedef __attribute__((ext_vector_type(4)))  int      v4i;

#define B_DIM   64      // batch rows (M)
#define N_DIM   8192    // output features
#define K_DIM   8192    // input features
#define KN      8064    // normal (int8) columns
#define KO      128     // outlier (fp16) columns
#define NBLK    126     // KN / 64 scale blocks
#define KPANEL  128     // K staged in LDS per panel
#define NPANELS 63      // KN / KPANEL (63 = 7*9 = 3*21)
#define LDSS    136     // LDS row stride in halves (128 + 8 pad -> conflict-free)
#define NTILE   128     // N per workgroup (8 waves x 16)

#if defined(__HIP_DEVICE_COMPILE__) && \
    __has_builtin(__builtin_amdgcn_global_load_async_to_lds_b128)
#define HAVE_ASYNC 1
#else
#define HAVE_ASYNC 0
#endif

// ---------------------------------------------------------------------------
// Kernel 1: gather x columns into packed fp16 buffers (normal + outlier).
// ---------------------------------------------------------------------------
__global__ void gather_cols_kernel(const float* __restrict__ x,
                                   const int* __restrict__ nidx,
                                   const int* __restrict__ oidx,
                                   _Float16* __restrict__ xn,
                                   _Float16* __restrict__ xo) {
  int t = blockIdx.x * blockDim.x + threadIdx.x;  // 64*8192 threads total
  int b = t >> 13;
  int i = t & 8191;
  const float* xrow = x + (size_t)b * K_DIM;
  if (i < KN) {
    xn[(size_t)b * KN + i] = (_Float16)xrow[nidx[i]];
  } else {
    int j = i - KN;
    xo[(size_t)b * KO + j] = (_Float16)xrow[oidx[j]];
  }
}

// ---------------------------------------------------------------------------
// Staging helpers: copy a 64 x 128-half panel into a padded LDS buffer.
// Async path uses GLOBAL_LOAD_ASYNC_TO_LDS_B128 (ASYNCcnt); fallback is a
// plain b128 copy through VGPRs. 1024 16-byte chunks / 256 threads = 4 each.
// ---------------------------------------------------------------------------
__device__ __forceinline__ void stage_panel(const _Float16* __restrict__ src,
                                            int srcStride, int k0,
                                            _Float16* dst, int tid) {
#pragma unroll
  for (int it = 0; it < 4; ++it) {
    int c = tid + it * 256;
    int row = c >> 4, cc = c & 15;
    const _Float16* g = &src[(size_t)row * srcStride + k0 + cc * 8];
    _Float16* l = &dst[row * LDSS + cc * 8];
#if HAVE_ASYNC
    __builtin_amdgcn_global_load_async_to_lds_b128(
        (__attribute__((address_space(1))) v4i*)g,
        (__attribute__((address_space(3))) v4i*)l, 0, 0);
#else
    *(v8h*)l = *(const v8h*)g;
#endif
  }
}

__device__ __forceinline__ void stage_join() {
#if HAVE_ASYNC
#if __has_builtin(__builtin_amdgcn_s_wait_asynccnt)
  __builtin_amdgcn_s_wait_asynccnt(0);
#else
  asm volatile("s_wait_asynccnt 0x0" ::: "memory");
#endif
#endif
  __syncthreads();
}

// ---------------------------------------------------------------------------
// Kernel 2: fused dequant + WMMA GEMM, K-split for latency hiding.
// grid = (N_DIM/NTILE) x KSPLIT workgroups of 8 waves. Each wave owns a
// 16-wide N slice and all four 16-row M tiles (M = 64). Split s handles
// panels [s*psplit, (s+1)*psplit); split 0 also does bias + fp16 outliers.
// Double-buffered LDS: stage panel p+1 while computing panel p.
// ---------------------------------------------------------------------------
__global__ __launch_bounds__(256, 1) void colout_gemm_kernel(
    const _Float16* __restrict__ xn,     // [64][8064] packed normal x (fp16)
    const _Float16* __restrict__ xo,     // [64][128]  packed outlier x (fp16)
    const int8_t*   __restrict__ qw,     // [8192][8064] companded int8 weights
    const float*    __restrict__ scales, // [8192][126]
    const _Float16* __restrict__ wo,     // [8192][128] outlier fp16 weights
    const float*    __restrict__ bias,   // [8192]
    float*          __restrict__ partials, // [KSPLIT][64][8192] f32
    int psplit) {                        // panels per split
  __shared__ _Float16 sx[2][B_DIM * LDSS];

  const int tid  = threadIdx.x;
  const int lane = tid & 31;
  const int wave = tid >> 5;
  const int nsub = lane & 15;  // N within the wave's 16-wide slice
  const int kh   = lane >> 4;  // K-half select (lanes 0-15 vs 16-31)
  const int n    = blockIdx.x * NTILE + wave * 16 + nsub;
  const int split  = blockIdx.y;
  const int pbegin = split * psplit;
  const int pend   = pbegin + psplit;

  v8f acc[4];
  int cur;  // LDS buffer holding the panel about to be computed

  if (split == 0) {  // uniform per block -> EXEC stays all-ones
    const float bv = bias[n];
#pragma unroll
    for (int m = 0; m < 4; ++m)
#pragma unroll
      for (int r = 0; r < 8; ++r) acc[m][r] = bv;

    // Stage outlier x into buf0 and the first int8 panel into buf1, then
    // run the fp16 outlier GEMM (K = 128) while nothing else is pending.
    stage_panel(xo, KO, 0, sx[0], tid);
    stage_panel(xn, KN, pbegin * KPANEL, sx[1], tid);
    stage_join();
#pragma unroll
    for (int kk = 0; kk < KO; kk += 32) {
      // B (32x16): each lane holds 16 contiguous K values of its weight row.
      const _Float16* wp = wo + (size_t)n * KO + kk + kh * 16;
      v8h b0 = *(const v8h*)(wp);
      v8h b1 = *(const v8h*)(wp + 8);
      v16h bmat = __builtin_shufflevector(b0, b1, 0, 1, 2, 3, 4, 5, 6, 7,
                                          8, 9, 10, 11, 12, 13, 14, 15);
#pragma unroll
      for (int m = 0; m < 4; ++m) {
        const _Float16* ap = &sx[0][(m * 16 + nsub) * LDSS + kk + kh * 16];
        v8h a0 = *(const v8h*)(ap);        // K = kk + kh*16 .. +7
        v8h a1 = *(const v8h*)(ap + 8);    // K = kk + kh*16 + 8 .. +15
        v16h amat = __builtin_shufflevector(a0, a1, 0, 1, 2, 3, 4, 5, 6, 7,
                                            8, 9, 10, 11, 12, 13, 14, 15);
        acc[m] = __builtin_amdgcn_wmma_f32_16x16x32_f16(
            false, amat, false, bmat, (short)0, acc[m], false, false);
      }
    }
    __syncthreads();  // buf0 reads done before main loop re-stages into it
    cur = 1;
  } else {
#pragma unroll
    for (int m = 0; m < 4; ++m)
#pragma unroll
      for (int r = 0; r < 8; ++r) acc[m][r] = 0.0f;
    stage_panel(xn, KN, pbegin * KPANEL, sx[0], tid);
    stage_join();
    cur = 0;
  }

  // -------- int8 companded path over this split's K panels --------
  const int8_t* qrow = qw + (size_t)n * KN;
  const float*  srow = scales + (size_t)n * NBLK;

  for (int p = pbegin; p < pend; ++p) {
    const int k0 = p * KPANEL;
    // Kick off the next panel's stage into the other buffer (overlapped with
    // this panel's WMMAs), and prefetch next weight cachelines into L2.
    if (p + 1 < pend)  // uniform branch
      stage_panel(xn, KN, k0 + KPANEL, sx[cur ^ 1], tid);
    __builtin_prefetch(qrow + k0 + KPANEL + kh * 16, 0, 3);

    const _Float16* sbuf = sx[cur];
#pragma unroll
    for (int ks = 0; ks < KPANEL; ks += 32) {
      const int kg = k0 + ks;
      // One b128 of int8 weights per lane -> 16 K values for (n, kh-half).
      union { v4i v; int8_t c[16]; } w;
      w.v = *(const v4i*)(qrow + kg + kh * 16);
      // Inverse companding: w = q*|q| * scale / 127^2  (block = kg/64).
      const float s = srow[kg >> 6] * (1.0f / 16129.0f);
      v16h bmat;
#pragma unroll
      for (int e = 0; e < 16; ++e) {
        float f = (float)(int)w.c[e];
        bmat[e] = (_Float16)(f * __builtin_fabsf(f) * s);
      }
#pragma unroll
      for (int m = 0; m < 4; ++m) {
        const _Float16* ap = &sbuf[(m * 16 + nsub) * LDSS + ks + kh * 16];
        v8h a0 = *(const v8h*)(ap);
        v8h a1 = *(const v8h*)(ap + 8);
        v16h amat = __builtin_shufflevector(a0, a1, 0, 1, 2, 3, 4, 5, 6, 7,
                                            8, 9, 10, 11, 12, 13, 14, 15);
        acc[m] = __builtin_amdgcn_wmma_f32_16x16x32_f16(
            false, amat, false, bmat, (short)0, acc[m], false, false);
      }
    }
    stage_join();  // next panel fully in LDS across all waves
    cur ^= 1;
  }

  // -------- write fp32 partial tile for this split --------
  // C/D layout: lane L, VGPR r -> M = (L>>4)*8 + r, N = L&15.
  float* pdst = partials + (size_t)split * B_DIM * N_DIM;
#pragma unroll
  for (int m = 0; m < 4; ++m) {
#pragma unroll
    for (int r = 0; r < 8; ++r) {
      pdst[(size_t)(m * 16 + kh * 8 + r) * N_DIM + n] = acc[m][r];
    }
  }
}

// ---------------------------------------------------------------------------
// Kernel 3: deterministic reduction of K-split partials -> fp16 output.
// Partials were just written, so they are served from the 192 MB L2.
// ---------------------------------------------------------------------------
__global__ void finalize_kernel(const float* __restrict__ partials,
                                _Float16* __restrict__ out, int nsplit) {
  int t = blockIdx.x * blockDim.x + threadIdx.x;  // (64*8192)/4 threads
  v4f s = {};
  for (int p = 0; p < nsplit; ++p)
    s += *(const v4f*)&partials[(size_t)p * B_DIM * N_DIM + t * 4];
  v4h h;
#pragma unroll
  for (int e = 0; e < 4; ++e) h[e] = (_Float16)s[e];
  *(v4h*)&out[t * 4] = h;
}

// ---------------------------------------------------------------------------
extern "C" void kernel_launch(void* const* d_in, const int* in_sizes, int n_in,
                              void* d_out, int out_size, void* d_ws, size_t ws_size,
                              hipStream_t stream) {
  (void)in_sizes; (void)n_in; (void)out_size;
  const float*    x      = (const float*)d_in[0];
  const int8_t*   qw     = (const int8_t*)d_in[1];
  const float*    scales = (const float*)d_in[2];
  const _Float16* wfp16  = (const _Float16*)d_in[3];
  const float*    bias   = (const float*)d_in[4];
  const int*      nidx   = (const int*)d_in[5];
  const int*      oidx   = (const int*)d_in[6];

  _Float16* xn  = (_Float16*)d_ws;                 // 64*8064 halves
  _Float16* xo  = xn + (size_t)B_DIM * KN;         // 64*128 halves
  float* partials = (float*)((char*)d_ws + (size_t)B_DIM * K_DIM * 2);
  _Float16* out = (_Float16*)d_out;                // 64*8192 halves

  // Choose K-split from available workspace (ws_size is fixed per run, so
  // this choice is deterministic). Need 1 MB gather + split*2 MB partials.
  const size_t base  = (size_t)B_DIM * K_DIM * 2;
  const size_t ptile = (size_t)B_DIM * N_DIM * 4;
  int ksplit = 7;                                  // 448 WGs -> 3584 waves
  if (ws_size < base + 7 * ptile) ksplit = 3;
  if (ws_size < base + 3 * ptile) ksplit = 1;
  const int psplit = NPANELS / ksplit;

  // 1) gather/convert x columns
  gather_cols_kernel<<<(B_DIM * K_DIM) / 256, 256, 0, stream>>>(x, nidx, oidx, xn, xo);
  // 2) fused dequant + WMMA GEMM (+outliers, +bias in split 0)
  dim3 grid(N_DIM / NTILE, ksplit);
  colout_gemm_kernel<<<grid, 256, 0, stream>>>(xn, xo, qw, scales, wfp16, bias,
                                               partials, psplit);
  // 3) deterministic partial reduction -> fp16
  finalize_kernel<<<(B_DIM * N_DIM) / (256 * 4), 256, 0, stream>>>(partials, out, ksplit);
}
